// MACCell_67396626809274
// MI455X (gfx1250) — matrix operand order repaired and checked by reading, not compiled, not run on hardware
//
#include <hip/hip_runtime.h>
#include <hip/hip_bf16.h>

// ---------------------------------------------------------------------------
// MAC cell for gfx1250 (MI455X). Heavy GEMMs via v_wmma_f32_16x16x32_bf16
// (f32 accumulate), double-buffered LDS, async global->LDS B-tile staging
// (global_load_async_to_lds_b128 / s_wait_asynccnt), weights pre-transposed
// to bf16 N x K once per call.
// ---------------------------------------------------------------------------

#define B_    32
#define QL_   128
#define DL_   1024
#define DIM_  512
#define NEGBIG 1e30f

typedef __attribute__((ext_vector_type(16))) __bf16          bf16x16;
typedef __attribute__((ext_vector_type(8)))  float           f32x8;
typedef __attribute__((ext_vector_type(8)))  unsigned short  u16x8;
typedef __attribute__((ext_vector_type(4)))  unsigned short  u16x4;

union FragB { bf16x16 v; u16x8 h[2]; };

enum { AM_PLAIN = 0, AM_CC = 1, AM_SCALE = 2, AM_BI = 3, AM_BF16A = 4 };

struct GemmP {
  const float*  A;    int lda;  long long aStride;   // f32 A modes
  const __bf16* A16;                                  // AM_BF16A (same lda/aStride)
  const __bf16* Bp16; int ldb;  long long bStride;   // bf16 N x K (transposed)
  float*        C;    int ldc;  long long cStride;
  const float*  bias;
  int M, N, K;
  int elu;
  // fused extras
  const float* pm;    // [B,512]  (AM_CC)
  const float* pk;    // [B*DL,512] (AM_CC)
  const float* cc;    // (AM_BI)
  const float* out1;  // (AM_BI)
  const float* out2;  // [B,512] (AM_BI)
  const float* scale; // [512] (AM_SCALE)
};

__device__ __forceinline__ float4 ld4(const float* p) { return *(const float4*)p; }

__device__ __forceinline__ void pk_store(__bf16* dst, float4 v) {
  union { u16x4 u; __bf16 b[4]; } t;
  t.b[0] = (__bf16)v.x; t.b[1] = (__bf16)v.y;
  t.b[2] = (__bf16)v.z; t.b[3] = (__bf16)v.w;
  *(u16x4*)dst = t.u;   // single ds_store_b64
}

// --- CDNA5 async global->LDS copy (16 bytes per lane), ASYNCcnt tracked ----
typedef __attribute__((address_space(3))) unsigned char lds_byte_t;

__device__ __forceinline__ void async_copy_b128(const void* gsrc, void* ldst) {
  unsigned loff = (unsigned)(unsigned long long)(lds_byte_t*)ldst;  // LDS byte offset
  asm volatile("global_load_async_to_lds_b128 %0, %1, off"
               :: "v"(loff), "v"(gsrc) : "memory");
}

__device__ __forceinline__ void wait_async0() {
  asm volatile("s_wait_asynccnt 0x0" ::: "memory");
}

template <int AMODE>
__device__ __forceinline__ float4 fetchA4(const GemmP& p, int bz, int rg, int kg) {
  if constexpr (AMODE == AM_PLAIN) {
    return ld4(&p.A[(size_t)bz * p.aStride + (size_t)rg * p.lda + kg]);
  } else if constexpr (AMODE == AM_CC) {
    const int bidx = rg >> 10;  // rows-per-batch = DL_
    if (kg < 512) {
      float4 a = ld4(&p.pk[(size_t)rg * 512 + kg]);
      float4 m = ld4(&p.pm[bidx * 512 + kg]);
      return make_float4(a.x * m.x, a.y * m.y, a.z * m.z, a.w * m.w);
    }
    return ld4(&p.pk[(size_t)rg * 512 + (kg - 512)]);
  } else if constexpr (AMODE == AM_SCALE) {
    float4 a = ld4(&p.A[(size_t)bz * p.aStride + (size_t)rg * p.lda + kg]);
    float4 s = ld4(&p.scale[kg]);
    return make_float4(a.x * s.x, a.y * s.y, a.z * s.z, a.w * s.w);
  } else {  // AM_BI : bi = [cc, out1, cc*out1, out2*out1]
    const size_t ro = (size_t)rg * 512;
    const int bidx  = rg >> 10;
    if (kg < 512)  return ld4(&p.cc[ro + kg]);
    if (kg < 1024) return ld4(&p.out1[ro + (kg - 512)]);
    if (kg < 1536) {
      float4 a = ld4(&p.cc[ro + (kg - 1024)]);
      float4 b = ld4(&p.out1[ro + (kg - 1024)]);
      return make_float4(a.x * b.x, a.y * b.y, a.z * b.z, a.w * b.w);
    }
    float4 a = ld4(&p.out2[bidx * 512 + (kg - 1536)]);
    float4 b = ld4(&p.out1[ro + (kg - 1536)]);
    return make_float4(a.x * b.x, a.y * b.y, a.z * b.z, a.w * b.w);
  }
}

// ------------------------------ WMMA GEMM ----------------------------------
// Block tile 128(M) x 64(N), BK=32. 256 threads = 8 waves (4Mx2N), each wave
// 32x32 via 2x2 WMMA 16x16x32 bf16. B tiles (bf16 NxK) staged with async
// global->LDS; f32 A modes staged via b128 loads + cvt_pk + ds_store_b64.

template <int AMODE>
__global__ __launch_bounds__(256) void wmma_gemm_kernel(GemmP p) {
  const int tid  = threadIdx.x;
  const int bn   = blockIdx.x, bm = blockIdx.y, bz = blockIdx.z;
  const int lane = tid & 31, wave = tid >> 5;
  const int wm   = wave & 3, wn = wave >> 2;

  __shared__ alignas(16) __bf16 sA[2][128 * 40];
  __shared__ alignas(16) __bf16 sB[2][64 * 40];

  f32x8 acc[2][2] = {};

  const int row0 = bm * 128;
  const int col0 = bn * 64;

  constexpr bool A_F32 = (AMODE != AM_BF16A);

  // staging thread layouts
  const int arow = tid >> 3;          // f32 A: 0..31, rows step +32
  const int ac4  = (tid & 7) << 2;
  const int ar2  = tid >> 1;          // bf16 A async: 0..127
  const int ac8  = (tid & 1) << 3;    // 0 / 8 (plus +16 second copy)
  const int brow = tid >> 2;          // B async: 0..63
  const int bc8  = (tid & 3) << 3;    // 0,8,16,24

  float4 ra[4];

  auto fetchA = [&](int k0) {
    if constexpr (A_F32) {
#pragma unroll
      for (int rr = 0; rr < 4; ++rr)
        ra[rr] = fetchA4<AMODE>(p, bz, row0 + arow + rr * 32, k0 + ac4);
    }
  };

  auto asyncAB = [&](int k0, int buf) {
    if constexpr (!A_F32) {
      const __bf16* src = &p.A16[(size_t)bz * p.aStride + (size_t)(row0 + ar2) * p.lda + k0];
      async_copy_b128(src + ac8,      &sA[buf][ar2 * 40 + ac8]);
      async_copy_b128(src + ac8 + 16, &sA[buf][ar2 * 40 + ac8 + 16]);
    }
    async_copy_b128(&p.Bp16[(size_t)bz * p.bStride + (size_t)(col0 + brow) * p.ldb + k0 + bc8],
                    &sB[buf][brow * 40 + bc8]);
  };

  auto stageA = [&](int buf) {
    if constexpr (A_F32) {
#pragma unroll
      for (int rr = 0; rr < 4; ++rr)
        pk_store(&sA[buf][(arow + rr * 32) * 40 + ac4], ra[rr]);
    }
  };

  const int hlf = lane >> 4;   // 0/1
  const int l16 = lane & 15;

  auto compute = [&](int buf) {
    FragB fa[2], fb[2];
#pragma unroll
    for (int mt = 0; mt < 2; ++mt) {
      const __bf16* pa = &sA[buf][(wm * 32 + mt * 16 + l16) * 40 + hlf * 8];
      fa[mt].h[0] = *(const u16x8*)pa;         // k = hlf*8 .. +7
      fa[mt].h[1] = *(const u16x8*)(pa + 16);  // k = 16+hlf*8 .. +7
    }
#pragma unroll
    for (int nt = 0; nt < 2; ++nt) {
      const __bf16* pb = &sB[buf][(wn * 32 + nt * 16 + l16) * 40 + hlf * 8];
      fb[nt].h[0] = *(const u16x8*)pb;         // k = hlf*8 .. +7
      fb[nt].h[1] = *(const u16x8*)(pb + 16);  // k = 16+hlf*8 .. +7
    }
#pragma unroll
    for (int mt = 0; mt < 2; ++mt)
#pragma unroll
      for (int nt = 0; nt < 2; ++nt)
        acc[mt][nt] = __builtin_amdgcn_wmma_f32_16x16x32_bf16(
            false, fa[mt].v, false, fb[nt].v, (short)0, acc[mt][nt], false, false);
  };

  const int nT = p.K >> 5;
  fetchA(0);
  asyncAB(0, 0);
  stageA(0);
  int cur = 0;
  for (int t = 0; t < nT; ++t) {
    wait_async0();                            // async tiles for 'cur' complete
    __syncthreads();
    const bool more = (t + 1 < nT);
    if (more) {
      fetchA((t + 1) << 5);                   // f32 b128 loads issued early
      asyncAB((t + 1) << 5, cur ^ 1);         // async copies into other buffer
    }
    if constexpr (AMODE == AM_PLAIN || AMODE == AM_SCALE) {
      if (t + 2 < nT)                         // global_prefetch_b8 for t+2
        __builtin_prefetch(&p.A[(size_t)bz * p.aStride +
                                (size_t)(row0 + arow) * p.lda + ((t + 2) << 5)], 0, 0);
    }
    compute(cur);                             // ds_load_b128 + v_wmma
    if (more) stageA(cur ^ 1);                // cvt_pk + ds_store_b64
    cur ^= 1;
  }

  // ---- epilogue ----
#pragma unroll
  for (int mt = 0; mt < 2; ++mt)
#pragma unroll
    for (int nt = 0; nt < 2; ++nt) {
      const int col = col0 + wn * 32 + nt * 16 + l16;
      const float bv = p.bias ? p.bias[col] : 0.f;
#pragma unroll
      for (int r = 0; r < 8; ++r) {
        const int row = row0 + wm * 32 + mt * 16 + hlf * 8 + r;
        float v = acc[mt][nt][r] + bv;
        if (p.elu) v = (v > 0.f) ? v : (__expf(v) - 1.f);
        p.C[(size_t)bz * p.cStride + (size_t)row * p.ldc + col] = v;
      }
    }
}

// --------------------- bf16 conversion / transpose -------------------------

__global__ void cvt_bf16_kernel(const float* __restrict__ in, __bf16* __restrict__ out,
                                long long n) {
  const long long i = (long long)blockIdx.x * blockDim.x + threadIdx.x;
  if (i < n) out[i] = (__bf16)in[i];
}

// per-batch transpose: in [batch][K][N] f32 -> out [batch][N][K] bf16
__global__ void cvtT_bf16_kernel(const float* __restrict__ in, __bf16* __restrict__ out,
                                 int K, int N) {
  const int b = blockIdx.z;
  const int idx = blockIdx.x * blockDim.x + threadIdx.x;
  if (idx >= K * N) return;
  const int n = idx / K, k = idx % K;  // write-coalesced
  out[((size_t)b * N + n) * K + k] = (__bf16)in[((size_t)b * K + k) * N + n];
}

// ------------------------- small helper kernels ----------------------------

__global__ void linear3_kernel(const float* A1, int K1, const float* A2, int K2,
                               const float* A3, int K3, const float* __restrict__ W,
                               const float* __restrict__ bias, float* __restrict__ C,
                               int M, int N, int act) {
  const int idx = blockIdx.x * blockDim.x + threadIdx.x;
  if (idx >= M * N) return;
  const int m = idx / N, n = idx % N;
  float s = bias ? bias[n] : 0.f;
  int koff = 0;
  for (int k = 0; k < K1; ++k) s += A1[(size_t)m * K1 + k] * W[(size_t)(koff + k) * N + n];
  koff += K1;
  for (int k = 0; k < K2; ++k) s += A2[(size_t)m * K2 + k] * W[(size_t)(koff + k) * N + n];
  koff += K2;
  for (int k = 0; k < K3; ++k) s += A3[(size_t)m * K3 + k] * W[(size_t)(koff + k) * N + n];
  if (act == 1) s = tanhf(s);
  C[idx] = s;
}

__global__ void ctrl_attn_kernel(const float* __restrict__ words, const float* __restrict__ cq,
                                 const float* __restrict__ Wc, const float* __restrict__ bc,
                                 const float* __restrict__ qmask, float* __restrict__ ctrl_attn,
                                 float* __restrict__ control) {
  const int b = blockIdx.x, q = threadIdx.x;
  __shared__ float sh[QL_];
  __shared__ float attn[QL_];
  const float* wrow = words + ((size_t)b * QL_ + q) * DIM_;
  const float* cqb  = cq + (size_t)b * DIM_;
  float s = 0.f;
  for (int d = 0; d < DIM_; ++d) s += cqb[d] * wrow[d] * Wc[d];
  s += bc[0];
  s -= NEGBIG * (1.f - qmask[b * QL_ + q]);
  sh[q] = s; __syncthreads();
  for (int off = QL_ / 2; off > 0; off >>= 1) { if (q < off) sh[q] = fmaxf(sh[q], sh[q + off]); __syncthreads(); }
  const float mx = sh[0]; __syncthreads();
  const float e = __expf(s - mx);
  sh[q] = e; __syncthreads();
  for (int off = QL_ / 2; off > 0; off >>= 1) { if (q < off) sh[q] += sh[q + off]; __syncthreads(); }
  const float a = e / sh[0];
  attn[q] = a;
  ctrl_attn[b * QL_ + q] = a;
  __syncthreads();
  for (int d = q; d < DIM_; d += QL_) {
    float acc = 0.f;
    for (int qq = 0; qq < QL_; ++qq) acc += attn[qq] * words[((size_t)b * QL_ + qq) * DIM_ + d];
    control[(size_t)b * DIM_ + d] = acc;
  }
}

__global__ void memdot_kernel(const float* __restrict__ words, const float* __restrict__ wbm,
                              float* __restrict__ mem_dot) {
  const int b = blockIdx.x, q = threadIdx.x;
  const float* wrow = words + ((size_t)b * QL_ + q) * DIM_;
  float s = 0.f;
  for (int d = 0; d < DIM_; ++d) s += wrow[d] * wbm[d];
  mem_dot[b * QL_ + q] = s;
}

// att assembly, softmax over q -> w1 (bf16), and s[b,i] = att . ctrl_attn
__global__ void att_kernel(const float* __restrict__ cross, const float* __restrict__ cc,
                           const float* __restrict__ wbi, const float* __restrict__ mem_dot,
                           const float* __restrict__ qmask, const float* __restrict__ ctrl_attn,
                           __bf16* __restrict__ w1b, float* __restrict__ s_out) {
  const int b = blockIdx.y, i = blockIdx.x, q = threadIdx.x;
  const size_t row = (size_t)b * DL_ + i;
  __shared__ float sh[QL_];
  float p = 0.f;
  for (int d = q; d < DIM_; d += QL_) p += cc[row * DIM_ + d] * wbi[d];
  sh[q] = p; __syncthreads();
  for (int off = QL_ / 2; off > 0; off >>= 1) { if (q < off) sh[q] += sh[q + off]; __syncthreads(); }
  const float indot = sh[0]; __syncthreads();
  const float a = cross[row * QL_ + q] + indot + mem_dot[b * QL_ + q] -
                  NEGBIG * (1.f - qmask[b * QL_ + q]);
  sh[q] = a * ctrl_attn[b * QL_ + q]; __syncthreads();
  for (int off = QL_ / 2; off > 0; off >>= 1) { if (q < off) sh[q] += sh[q + off]; __syncthreads(); }
  if (q == 0) s_out[row] = sh[0];
  __syncthreads();
  sh[q] = a; __syncthreads();
  for (int off = QL_ / 2; off > 0; off >>= 1) { if (q < off) sh[q] = fmaxf(sh[q], sh[q + off]); __syncthreads(); }
  const float mx = sh[0]; __syncthreads();
  const float e = __expf(a - mx);
  sh[q] = e; __syncthreads();
  for (int off = QL_ / 2; off > 0; off >>= 1) { if (q < off) sh[q] += sh[q + off]; __syncthreads(); }
  w1b[row * QL_ + q] = (__bf16)(e / sh[0]);
}

__global__ void w2out2_kernel(const float* __restrict__ s_in, const float* __restrict__ cc,
                              float* __restrict__ out2) {
  const int b = blockIdx.x, t = threadIdx.x;
  __shared__ float w2s[DL_];
  __shared__ float red[256];
  float mx = -3.4e38f;
  for (int i = t; i < DL_; i += 256) mx = fmaxf(mx, s_in[(size_t)b * DL_ + i]);
  red[t] = mx; __syncthreads();
  for (int off = 128; off > 0; off >>= 1) { if (t < off) red[t] = fmaxf(red[t], red[t + off]); __syncthreads(); }
  mx = red[0]; __syncthreads();
  float sum = 0.f;
  for (int i = t; i < DL_; i += 256) { const float e = __expf(s_in[(size_t)b * DL_ + i] - mx); w2s[i] = e; sum += e; }
  red[t] = sum; __syncthreads();
  for (int off = 128; off > 0; off >>= 1) { if (t < off) red[t] += red[t + off]; __syncthreads(); }
  const float tot = red[0]; __syncthreads();
  for (int i = t; i < DL_; i += 256) w2s[i] /= tot;
  __syncthreads();
  for (int d = t; d < DIM_; d += 256) {
    float acc = 0.f;
    for (int i = 0; i < DL_; ++i) acc += w2s[i] * cc[((size_t)b * DL_ + i) * DIM_ + d];
    out2[(size_t)b * DIM_ + d] = acc;
  }
}

__global__ void read_kernel(const float* __restrict__ out, const float* __restrict__ Wr,
                            const float* __restrict__ br, const float* __restrict__ dmask,
                            const float* __restrict__ img, float* __restrict__ readv) {
  const int b = blockIdx.x, t = threadIdx.x;
  __shared__ float ra[DL_];
  __shared__ float red[256];
  for (int i = t; i < DL_; i += 256) {
    const float* orow = out + ((size_t)b * DL_ + i) * DIM_;
    float s = 0.f;
    for (int d = 0; d < DIM_; ++d) s += orow[d] * Wr[d];
    s += br[0] - NEGBIG * (1.f - dmask[b * DL_ + i]);
    ra[i] = s;
  }
  __syncthreads();
  float mx = -3.4e38f;
  for (int i = t; i < DL_; i += 256) mx = fmaxf(mx, ra[i]);
  red[t] = mx; __syncthreads();
  for (int off = 128; off > 0; off >>= 1) { if (t < off) red[t] = fmaxf(red[t], red[t + off]); __syncthreads(); }
  mx = red[0]; __syncthreads();
  float sum = 0.f;
  for (int i = t; i < DL_; i += 256) { const float e = __expf(ra[i] - mx); ra[i] = e; sum += e; }
  red[t] = sum; __syncthreads();
  for (int off = 128; off > 0; off >>= 1) { if (t < off) red[t] += red[t + off]; __syncthreads(); }
  const float tot = red[0]; __syncthreads();
  for (int i = t; i < DL_; i += 256) ra[i] /= tot;
  __syncthreads();
  for (int d = t; d < DIM_; d += 256) {
    float acc = 0.f;
    for (int i = 0; i < DL_; ++i) acc += ra[i] * img[((size_t)b * DL_ + i) * DIM_ + d];
    readv[(size_t)b * DIM_ + d] = acc;
  }
}

// ------------------------------- launch ------------------------------------

extern "C" void kernel_launch(void* const* d_in, const int* in_sizes, int n_in,
                              void* d_out, int out_size, void* d_ws, size_t ws_size,
                              hipStream_t stream) {
  const float* words    = (const float*)d_in[0];
  const float* question = (const float*)d_in[1];
  const float* img      = (const float*)d_in[2];
  const float* qmask    = (const float*)d_in[3];
  const float* dmask    = (const float*)d_in[4];
  const float* memory   = (const float*)d_in[5];
  const float* W_scp = (const float*)d_in[6],  *b_scp = (const float*)d_in[7];
  const float* W_pa  = (const float*)d_in[8],  *b_pa  = (const float*)d_in[9];
  const float* W_cq  = (const float*)d_in[10], *b_cq  = (const float*)d_in[11];
  const float* W_cattn = (const float*)d_in[12], *b_cattn = (const float*)d_in[13];
  const float* W_mem = (const float*)d_in[14], *b_mem = (const float*)d_in[15];
  const float* W_kb  = (const float*)d_in[16], *b_kb  = (const float*)d_in[17];
  const float* W_cat = (const float*)d_in[18], *b_cat = (const float*)d_in[19];
  const float* W_rattn = (const float*)d_in[20], *b_rattn = (const float*)d_in[21];
  const float* W_lin = (const float*)d_in[22], *b_lin = (const float*)d_in[23];
  const float* w_bi_in  = (const float*)d_in[24];
  const float* w_bi_mem = (const float*)d_in[25];
  const float* dot_scale = (const float*)d_in[26];
  const float* W_write = (const float*)d_in[27], *b_write = (const float*)d_in[28];

  float* out_o     = (float*)d_out;
  float* nextmem_o = out_o + (size_t)B_ * DL_ * DIM_;
  float* control_o = nextmem_o + (size_t)B_ * DIM_;

  // -------- workspace layout (float units; bf16 regions cast) --------
  float* ws = (float*)d_ws;
  float* q_     = ws;
  float* pa_    = q_    + (size_t)B_ * DIM_;
  float* cq_    = pa_   + (size_t)B_ * DIM_;
  float* ctrlA  = cq_   + (size_t)B_ * DIM_;
  float* pm_    = ctrlA + (size_t)B_ * QL_;
  float* mdot_  = pm_   + (size_t)B_ * DIM_;
  float* s_     = mdot_ + (size_t)B_ * QL_;
  float* out2_  = s_    + (size_t)B_ * DL_;
  float* read_  = out2_ + (size_t)B_ * DIM_;
  float* pk_    = read_ + (size_t)B_ * DIM_;
  float* cc_    = pk_   + (size_t)B_ * DL_ * DIM_;
  float* cross_ = cc_   + (size_t)B_ * DL_ * DIM_;
  float* out1_  = cross_ + (size_t)B_ * DL_ * QL_;
  float* bf16h  = out1_ + (size_t)B_ * DL_ * DIM_;   // start of bf16 area
  __bf16* w1b     = (__bf16*)bf16h;                          // 32*1024*128
  __bf16* wordsBF = w1b     + (size_t)B_ * DL_ * QL_;        // 32*128*512 (plain)
  __bf16* wordsT  = wordsBF + (size_t)B_ * QL_ * DIM_;       // 32*512*128 (NxK)
  __bf16* wkbT    = wordsT  + (size_t)B_ * DIM_ * QL_;       // 512x512
  __bf16* wcatT   = wkbT    + (size_t)DIM_ * DIM_;           // 512x1024
  __bf16* wlinT   = wcatT   + (size_t)DIM_ * 2 * DIM_;       // 512x2048

  const int lin_blk = 256;
  const int lin_grid = (B_ * DIM_ + lin_blk - 1) / lin_blk;

  // -------- one-time bf16 packing (few MB, bandwidth-trivial) --------
  {
    long long n = (long long)B_ * QL_ * DIM_;
    cvt_bf16_kernel<<<(unsigned)((n + 255) / 256), 256, 0, stream>>>(words, wordsBF, n);
    cvtT_bf16_kernel<<<dim3((QL_ * DIM_ + 255) / 256, 1, B_), 256, 0, stream>>>(words, wordsT, QL_, DIM_);
    cvtT_bf16_kernel<<<dim3((DIM_ * DIM_ + 255) / 256, 1, 1), 256, 0, stream>>>(W_kb, wkbT, DIM_, DIM_);
    cvtT_bf16_kernel<<<dim3((2 * DIM_ * DIM_ + 255) / 256, 1, 1), 256, 0, stream>>>(W_cat, wcatT, 2 * DIM_, DIM_);
    cvtT_bf16_kernel<<<dim3((4 * DIM_ * DIM_ + 255) / 256, 1, 1), 256, 0, stream>>>(W_lin, wlinT, 4 * DIM_, DIM_);
  }

  // --- ControlUnit ---
  linear3_kernel<<<lin_grid, lin_blk, 0, stream>>>(question, DIM_, nullptr, 0, nullptr, 0,
                                                   W_scp, b_scp, q_, B_, DIM_, 1);
  linear3_kernel<<<lin_grid, lin_blk, 0, stream>>>(q_, DIM_, nullptr, 0, nullptr, 0,
                                                   W_pa, b_pa, pa_, B_, DIM_, 0);
  linear3_kernel<<<lin_grid, lin_blk, 0, stream>>>(question, DIM_, pa_, DIM_, nullptr, 0,
                                                   W_cq, b_cq, cq_, B_, DIM_, 0);
  ctrl_attn_kernel<<<B_, QL_, 0, stream>>>(words, cq_, W_cattn, b_cattn, qmask, ctrlA, control_o);

  // --- ReadUnit: pm, pk, cc ---
  linear3_kernel<<<lin_grid, lin_blk, 0, stream>>>(memory, DIM_, nullptr, 0, nullptr, 0,
                                                   W_mem, b_mem, pm_, B_, DIM_, 0);
  {
    GemmP p = {};
    p.A = img; p.lda = DIM_; p.aStride = 0;
    p.Bp16 = wkbT; p.ldb = DIM_; p.bStride = 0;
    p.C = pk_; p.ldc = DIM_; p.cStride = 0;
    p.bias = b_kb; p.M = B_ * DL_; p.N = DIM_; p.K = DIM_; p.elu = 0;
    wmma_gemm_kernel<AM_PLAIN>
        <<<dim3(DIM_ / 64, (B_ * DL_) / 128, 1), 256, 0, stream>>>(p);
  }
  {
    GemmP p = {};
    p.pm = pm_; p.pk = pk_;
    p.Bp16 = wcatT; p.ldb = 2 * DIM_; p.bStride = 0;
    p.C = cc_; p.ldc = DIM_; p.cStride = 0;
    p.bias = b_cat; p.M = B_ * DL_; p.N = DIM_; p.K = 2 * DIM_; p.elu = 0;
    wmma_gemm_kernel<AM_CC>
        <<<dim3(DIM_ / 64, (B_ * DL_) / 128, 1), 256, 0, stream>>>(p);
  }

  // --- CtrlBiAttention ---
  memdot_kernel<<<B_, QL_, 0, stream>>>(words, w_bi_mem, mdot_);
  {
    GemmP p = {};
    p.A = cc_; p.lda = DIM_; p.aStride = (long long)DL_ * DIM_;
    p.scale = dot_scale;
    p.Bp16 = wordsBF; p.ldb = DIM_; p.bStride = (long long)QL_ * DIM_;
    p.C = cross_; p.ldc = QL_; p.cStride = (long long)DL_ * QL_;
    p.bias = nullptr; p.M = DL_; p.N = QL_; p.K = DIM_; p.elu = 0;
    wmma_gemm_kernel<AM_SCALE>
        <<<dim3(QL_ / 64, DL_ / 128, B_), 256, 0, stream>>>(p);
  }
  att_kernel<<<dim3(DL_, B_), QL_, 0, stream>>>(cross_, cc_, w_bi_in, mdot_, qmask, ctrlA, w1b, s_);
  w2out2_kernel<<<B_, 256, 0, stream>>>(s_, cc_, out2_);
  {
    GemmP p = {};
    p.A16 = w1b; p.lda = QL_; p.aStride = (long long)DL_ * QL_;
    p.Bp16 = wordsT; p.ldb = QL_; p.bStride = (long long)DIM_ * QL_;
    p.C = out1_; p.ldc = DIM_; p.cStride = (long long)DL_ * DIM_;
    p.bias = nullptr; p.M = DL_; p.N = DIM_; p.K = QL_; p.elu = 0;
    wmma_gemm_kernel<AM_BF16A>
        <<<dim3(DIM_ / 64, DL_ / 128, B_), 256, 0, stream>>>(p);
  }

  // --- out = elu(bi @ W_lin + b_lin), bi fused in A-loader ---
  {
    GemmP p = {};
    p.cc = cc_; p.out1 = out1_; p.out2 = out2_;
    p.Bp16 = wlinT; p.ldb = 4 * DIM_; p.bStride = 0;
    p.C = out_o; p.ldc = DIM_; p.cStride = 0;
    p.bias = b_lin; p.M = B_ * DL_; p.N = DIM_; p.K = 4 * DIM_; p.elu = 1;
    wmma_gemm_kernel<AM_BI>
        <<<dim3(DIM_ / 64, (B_ * DL_) / 128, 1), 256, 0, stream>>>(p);
  }

  // --- read attention + WriteUnit ---
  read_kernel<<<B_, 256, 0, stream>>>(out_o, W_rattn, b_rattn, dmask, img, read_);
  linear3_kernel<<<lin_grid, lin_blk, 0, stream>>>(read_, DIM_, memory, DIM_, control_o, DIM_,
                                                   W_write, b_write, nextmem_o, B_, DIM_, 0);
}
